// SDPA_90366111908413
// MI455X (gfx1250) — matrix operand order repaired and checked by reading, not compiled, not run on hardware
//
#include <hip/hip_runtime.h>
#include <hip/hip_bf16.h>
#include <stdint.h>

// ---------------------------------------------------------------------------
// SDPA (causal, single head, d=1024) for gfx1250.
// Stage 1: fp32 -> bf16 conversion of x and weights.
// Stage 2: QKV projection GEMMs via v_wmma_f32_16x16x32_bf16 (TN layout,
//          both operands K-contiguous). V stored transposed for PV step.
// Stage 3: flash attention, 16 queries/block, 8 waves x 128 out-cols,
//          32-key tiles. K tiles staged into LDS by the Tensor Data Mover
//          (double-buffered, overlapped with WMMA compute), online softmax,
//          P repacked through LDS into A-fragment layout, PV via WMMA,
//          V^T prefetched with global_prefetch.
// ---------------------------------------------------------------------------

#define BATCH 4
#define SEQ   2048
#define DIM   1024   // D_IN == D_KQ == D_OUT

typedef __attribute__((ext_vector_type(16))) __bf16         v16bf;
typedef __attribute__((ext_vector_type(8)))  float          v8f;
typedef __attribute__((ext_vector_type(4)))  unsigned int   u32x4;
typedef __attribute__((ext_vector_type(8)))  int            i32x8;
typedef __attribute__((ext_vector_type(4)))  int            i32x4;
typedef __attribute__((ext_vector_type(4)))  float          f32x4;
typedef __attribute__((ext_vector_type(4)))  unsigned short u16x4;

#if defined(__has_builtin)
#  if __has_builtin(__builtin_amdgcn_tensor_load_to_lds)
#    define HAVE_TDM 1
#  else
#    define HAVE_TDM 0
#  endif
#  if __has_builtin(__builtin_amdgcn_s_wait_tensorcnt)
#    define HAVE_WAIT_TENSOR_BUILTIN 1
#  else
#    define HAVE_WAIT_TENSOR_BUILTIN 0
#  endif
#else
#  define HAVE_TDM 0
#  define HAVE_WAIT_TENSOR_BUILTIN 0
#endif

union FragBF {
    v16bf bf;
    u32x4 q[2];
};

__device__ __forceinline__ unsigned short f2bf(float f) {
    // round-to-nearest-even fp32 -> bf16 (raw bits)
    unsigned int u = __builtin_bit_cast(unsigned int, f);
    u += 0x7FFFu + ((u >> 16) & 1u);
    return (unsigned short)(u >> 16);
}

__device__ __forceinline__ v8f wmma_bf16(const FragBF& a, const FragBF& b, v8f c) {
    return __builtin_amdgcn_wmma_f32_16x16x32_bf16(
        false, a.bf, false, b.bf, (short)0, c, false, false);
}

__device__ __forceinline__ void wait_tensorcnt0() {
#if HAVE_WAIT_TENSOR_BUILTIN
    __builtin_amdgcn_s_wait_tensorcnt(0);
#else
    asm volatile("s_wait_tensorcnt 0x0" ::: "memory");
#endif
}

#if HAVE_TDM
// TDM: DMA one 32x1024 bf16 tile (row-major, stride 1024) from global to LDS.
// D# per CDNA5 ISA ch8 "Tensor DMA Descriptor".
__device__ __forceinline__ void tdm_load_k_tile(const unsigned short* gsrc,
                                                unsigned short* ldst) {
    unsigned long long ga = (unsigned long long)(uintptr_t)gsrc;
    unsigned int       la = (unsigned int)(uintptr_t)ldst; // low 32b = LDS offset

    u32x4 g0;
    g0[0] = 1u;                                   // count=1, is_restore=0
    g0[1] = la;                                   // lds_addr (bytes)
    g0[2] = (unsigned int)ga;                     // global_addr[31:0]
    g0[3] = (unsigned int)(ga >> 32) | 0x80000000u; // addr[56:32] | type=2

    i32x8 g1;
    g1[0] = 0x00010000;            // workgroup_mask=0, data_size=1 (2 bytes)
    g1[1] = (int)(1024u << 16);    // tensor_dim0 = 1024 (bits 79:48 lo half)
    g1[2] = (int)(2048u << 16);    // tensor_dim0 hi=0 | tensor_dim1 = 2048 lo
    g1[3] = (int)(1024u << 16);    // tensor_dim1 hi=0 | tile_dim0 = 1024
    g1[4] = 32;                    // tile_dim1 = 32, tile_dim2 = 0
    g1[5] = 1024;                  // tensor_dim0_stride = 1024
    g1[6] = 0;                     // stride hi / tensor_dim1_stride lo
    g1[7] = 0;

    i32x4 z4 = {0, 0, 0, 0};
#if __clang_major__ >= 23
    i32x8 z8 = {0, 0, 0, 0, 0, 0, 0, 0};
    __builtin_amdgcn_tensor_load_to_lds(g0, g1, z4, z4, z8, 0);
#else
    __builtin_amdgcn_tensor_load_to_lds(g0, g1, z4, z4, 0);
#endif
}
#endif

// ---------------------------------------------------------------------------
// Stage 1: fp32 -> raw bf16
// ---------------------------------------------------------------------------
__global__ void cvt_f32_to_bf16(const float* __restrict__ src,
                                unsigned short* __restrict__ dst, long n) {
    long i = ((long)blockIdx.x * blockDim.x + threadIdx.x) * 4;
    if (i + 3 < n) {
        f32x4 v = *(const f32x4*)(src + i);
        u16x4 r;
        r.x = f2bf(v.x); r.y = f2bf(v.y); r.z = f2bf(v.z); r.w = f2bf(v.w);
        *(u16x4*)(dst + i) = r;
    }
}

// ---------------------------------------------------------------------------
// Stage 2: y[m,n] = sum_k x[m,k] * W[n,k] + bias[n]
// grid = (N/64, M/64, 3)  z: 0=Q, 1=K, 2=V(transposed store)
// block = 128 threads = 4 waves, each wave computes 32x32
// ---------------------------------------------------------------------------
__global__ void __launch_bounds__(128)
qkv_gemm(const unsigned short* __restrict__ xb,
         const unsigned short* __restrict__ wq,
         const unsigned short* __restrict__ wk,
         const unsigned short* __restrict__ wv,
         const float* __restrict__ bq,
         const float* __restrict__ bk,
         const float* __restrict__ bv,
         unsigned short* __restrict__ outq,
         unsigned short* __restrict__ outk,
         unsigned short* __restrict__ outvt) {
    const int z = blockIdx.z;
    const unsigned short* W    = (z == 0) ? wq : (z == 1) ? wk : wv;
    const float*          bias = (z == 0) ? bq : (z == 1) ? bk : bv;
    unsigned short*       out  = (z == 0) ? outq : (z == 1) ? outk : outvt;

    const int lane = threadIdx.x & 31;
    const int wave = threadIdx.x >> 5;
    const int hl   = lane >> 4;      // which 16-lane half
    const int l16  = lane & 15;

    const int m0 = blockIdx.y * 64 + (wave >> 1) * 32;
    const int n0 = blockIdx.x * 64 + (wave & 1) * 32;

    v8f c[2][2] = {};

    for (int k0 = 0; k0 < DIM; k0 += 32) {
        FragBF a[2], b[2];
#pragma unroll
        for (int mf = 0; mf < 2; ++mf) {
            // A 16x32 bf16: lane row = l16, K = hl*16 + {0..15} (contiguous)
            const unsigned short* p =
                xb + (size_t)(m0 + mf * 16 + l16) * DIM + k0 + hl * 16;
            a[mf].q[0] = *(const u32x4*)(p);
            a[mf].q[1] = *(const u32x4*)(p + 8);
        }
#pragma unroll
        for (int nf = 0; nf < 2; ++nf) {
            // B 32x16 bf16: lane col = l16, K = hl*16 + {0..15} (contiguous)
            const unsigned short* p =
                W + (size_t)(n0 + nf * 16 + l16) * DIM + k0 + hl * 16;
            b[nf].q[0] = *(const u32x4*)(p);
            b[nf].q[1] = *(const u32x4*)(p + 8);
        }
#pragma unroll
        for (int mf = 0; mf < 2; ++mf)
#pragma unroll
            for (int nf = 0; nf < 2; ++nf)
                c[mf][nf] = wmma_bf16(a[mf], b[nf], c[mf][nf]);
    }

#pragma unroll
    for (int mf = 0; mf < 2; ++mf) {
#pragma unroll
        for (int nf = 0; nf < 2; ++nf) {
            const int col = n0 + nf * 16 + l16;
            const float bcol = bias[col];
            const int mbase = m0 + mf * 16 + hl * 8;
#pragma unroll
            for (int r = 0; r < 8; ++r) {
                const int m = mbase + r;
                const unsigned short h = f2bf(c[mf][nf][r] + bcol);
                if (z == 2) {
                    const int bb = m >> 11;       // m / SEQ
                    const int s  = m & (SEQ - 1); // m % SEQ
                    out[((size_t)bb * DIM + col) * SEQ + s] = h; // V^T
                } else {
                    out[(size_t)m * DIM + col] = h;
                }
            }
        }
    }
}

// ---------------------------------------------------------------------------
// Stage 3: flash attention.
// grid = (SEQ/16, BATCH), block = 256 (8 waves). Block owns 16 query rows;
// wave w owns output columns [w*128, w*128+128). 32-key tiles.
// LDS: sQ 16x1024 (32KB) + sK 2x32x1024 (128KB, TDM double-buffer)
//      + sP 8x16x32 (8KB) = 168KB dynamic.
// ---------------------------------------------------------------------------
__global__ void __launch_bounds__(256)
flash_attn(const unsigned short* __restrict__ Qb,
           const unsigned short* __restrict__ Kb,
           const unsigned short* __restrict__ Vtb,
           float* __restrict__ out) {
    extern __shared__ __align__(16) unsigned short sm[];
    unsigned short* sQ = sm;                       // [16][1024]
    unsigned short* sK = sm + 16 * DIM;            // 2 x [32][1024]
    unsigned short* sP = sm + 16 * DIM + 2 * 32 * DIM; // [8 waves][16][32]

    const int b   = blockIdx.y;
    const int q0  = blockIdx.x * 16;
    const int tid = threadIdx.x;
    const int lane = tid & 31;
    const int wave = tid >> 5;
    const int hl   = lane >> 4;
    const int l16  = lane & 15;

    const unsigned short* Kbase = Kb  + (size_t)b * SEQ * DIM;
    const unsigned short* Vbase = Vtb + (size_t)b * DIM * SEQ;
    unsigned short* myP = sP + wave * (16 * 32);
    const int col0 = wave * 128;
    const int ntiles = (q0 + 16 + 31) >> 5;

    // Stage Q tile (contiguous 32KB region), all threads
    {
        const u32x4* src = (const u32x4*)(Qb + ((size_t)b * SEQ + q0) * DIM);
        u32x4* dst = (u32x4*)sQ;
#pragma unroll
        for (int c = 0; c < (16 * DIM) / 8 / 256; ++c)
            dst[tid + c * 256] = src[tid + c * 256];
    }

    // Stage K tile 0 (TDM from wave 0, or cooperative fallback)
#if HAVE_TDM
    if (tid < 32) {
        tdm_load_k_tile(Kbase, sK);
        wait_tensorcnt0();
    }
#else
    for (int c = tid; c < (32 * DIM) / 8; c += 256)
        ((u32x4*)sK)[c] = ((const u32x4*)Kbase)[c];
#endif
    __syncthreads();

    v8f o[8] = {};
    float rmax[8], rsum[8];
#pragma unroll
    for (int r = 0; r < 8; ++r) { rmax[r] = -__builtin_inff(); rsum[r] = 0.f; }

    for (int t = 0; t < ntiles; ++t) {
        const int j0 = t * 32;
        const unsigned short* kbuf = sK + (t & 1) * (32 * DIM);

        // Kick off DMA of the next K tile into the other buffer
        if (t + 1 < ntiles) {
#if HAVE_TDM
            if (tid < 32)
                tdm_load_k_tile(Kbase + (size_t)(j0 + 32) * DIM,
                                sK + ((t + 1) & 1) * (32 * DIM));
#else
            unsigned short* dst = sK + ((t + 1) & 1) * (32 * DIM);
            const unsigned short* src = Kbase + (size_t)(j0 + 32) * DIM;
            for (int c = tid; c < (32 * DIM) / 8; c += 256)
                ((u32x4*)dst)[c] = ((const u32x4*)src)[c];
#endif
        }

        v8f s0 = {}, s1 = {};

        // S = Q * K^T over d=1024, two 16-key column tiles (K from LDS)
#pragma unroll 4
        for (int kk = 0; kk < DIM; kk += 32) {
            FragBF qa, kf0, kf1;
            const unsigned short* qp = sQ + l16 * DIM + kk + hl * 16;
            qa.q[0] = *(const u32x4*)(qp);
            qa.q[1] = *(const u32x4*)(qp + 8);
            const unsigned short* kp0 = kbuf + (size_t)l16 * DIM + kk + hl * 16;
            kf0.q[0] = *(const u32x4*)(kp0);
            kf0.q[1] = *(const u32x4*)(kp0 + 8);
            const unsigned short* kp1 =
                kbuf + (size_t)(16 + l16) * DIM + kk + hl * 16;
            kf1.q[0] = *(const u32x4*)(kp1);
            kf1.q[1] = *(const u32x4*)(kp1 + 8);
            s0 = wmma_bf16(qa, kf0, s0);
            s1 = wmma_bf16(qa, kf1, s1);
        }

        // online softmax (rows live across 16-lane groups; shuffle-reduce)
        const float sc = 0.03125f; // 1/sqrt(1024)
        float alpha[8];
#pragma unroll
        for (int r = 0; r < 8; ++r) {
            const int qrow = q0 + r + hl * 8;
            float e0 = s0[r] * sc;
            float e1 = s1[r] * sc;
            if (j0 + l16 > qrow)      e0 = -__builtin_inff();
            if (j0 + 16 + l16 > qrow) e1 = -__builtin_inff();
            float mx = fmaxf(e0, e1);
            mx = fmaxf(mx, __shfl_xor(mx, 1, 32));
            mx = fmaxf(mx, __shfl_xor(mx, 2, 32));
            mx = fmaxf(mx, __shfl_xor(mx, 4, 32));
            mx = fmaxf(mx, __shfl_xor(mx, 8, 32));
            const float mnew = fmaxf(rmax[r], mx);
            const float al = __expf(rmax[r] - mnew); // first tile: exp(-inf)=0
            const float p0 = __expf(e0 - mnew);
            const float p1 = __expf(e1 - mnew);
            float ps = p0 + p1;
            ps += __shfl_xor(ps, 1, 32);
            ps += __shfl_xor(ps, 2, 32);
            ps += __shfl_xor(ps, 4, 32);
            ps += __shfl_xor(ps, 8, 32);
            rsum[r] = rsum[r] * al + ps;
            rmax[r] = mnew;
            alpha[r] = al;
            // stage P (16x32 bf16, row-major) for A-fragment reload
            const int prow = r + hl * 8;
            myP[prow * 32 + l16]      = f2bf(p0);
            myP[prow * 32 + 16 + l16] = f2bf(p1);
        }

        // rescale running output accumulators
#pragma unroll
        for (int cf = 0; cf < 8; ++cf)
#pragma unroll
            for (int r = 0; r < 8; ++r)
                o[cf][r] *= alpha[r];

        // per-wave LDS store->load; DS ops from one wave execute in order
        __asm__ volatile("" ::: "memory");

        FragBF pa;
        const unsigned short* pp = myP + l16 * 32 + hl * 16;
        pa.q[0] = *(const u32x4*)(pp);
        pa.q[1] = *(const u32x4*)(pp + 8);

        // O += P(16x32) * V^T(32x128-col slice); prefetch next tile's V
#pragma unroll
        for (int cf = 0; cf < 8; ++cf) {
            FragBF vf;
            const unsigned short* vp =
                Vbase + (size_t)(col0 + cf * 16 + l16) * SEQ + j0 + hl * 16;
            vf.q[0] = *(const u32x4*)(vp);
            vf.q[1] = *(const u32x4*)(vp + 8);
            if (t + 1 < ntiles) __builtin_prefetch(vp + 32, 0, 0);
            o[cf] = wmma_bf16(pa, vf, o[cf]);
        }

        // Next K tile must be resident before anyone reads it
#if HAVE_TDM
        if (tid < 32) wait_tensorcnt0();
#endif
        __syncthreads();
    }

    // normalize + store fp32
#pragma unroll
    for (int cf = 0; cf < 8; ++cf) {
        const int col = col0 + cf * 16 + l16;
#pragma unroll
        for (int r = 0; r < 8; ++r) {
            const int qrow = q0 + r + hl * 8;
            out[((size_t)b * SEQ + qrow) * DIM + col] = o[cf][r] / rsum[r];
        }
    }
}

// ---------------------------------------------------------------------------
// Launcher
// ---------------------------------------------------------------------------
extern "C" void kernel_launch(void* const* d_in, const int* in_sizes, int n_in,
                              void* d_out, int out_size, void* d_ws, size_t ws_size,
                              hipStream_t stream) {
    const float* x  = (const float*)d_in[0];
    const float* Wq = (const float*)d_in[1];
    const float* bq = (const float*)d_in[2];
    const float* Wk = (const float*)d_in[3];
    const float* bk = (const float*)d_in[4];
    const float* Wv = (const float*)d_in[5];
    const float* bv = (const float*)d_in[6];
    float* out = (float*)d_out;

    // workspace layout (bytes)
    char* ws = (char*)d_ws;
    const size_t SZ_X = (size_t)BATCH * SEQ * DIM * 2;  // 16 MB
    const size_t SZ_W = (size_t)DIM * DIM * 2;          //  2 MB
    unsigned short* xb  = (unsigned short*)(ws);
    unsigned short* wqb = (unsigned short*)(ws + SZ_X);
    unsigned short* wkb = (unsigned short*)(ws + SZ_X + SZ_W);
    unsigned short* wvb = (unsigned short*)(ws + SZ_X + 2 * SZ_W);
    unsigned short* qb  = (unsigned short*)(ws + SZ_X + 3 * SZ_W);
    unsigned short* kb  = (unsigned short*)(ws + 2 * SZ_X + 3 * SZ_W);
    unsigned short* vtb = (unsigned short*)(ws + 3 * SZ_X + 3 * SZ_W);

    // Stage 1: conversions
    const long nx = (long)BATCH * SEQ * DIM;   // 8388608
    const long nw = (long)DIM * DIM;           // 1048576
    cvt_f32_to_bf16<<<(unsigned)(nx / 4 / 256), 256, 0, stream>>>(x,  xb,  nx);
    cvt_f32_to_bf16<<<(unsigned)(nw / 4 / 256), 256, 0, stream>>>(Wq, wqb, nw);
    cvt_f32_to_bf16<<<(unsigned)(nw / 4 / 256), 256, 0, stream>>>(Wk, wkb, nw);
    cvt_f32_to_bf16<<<(unsigned)(nw / 4 / 256), 256, 0, stream>>>(Wv, wvb, nw);

    // Stage 2: QKV projections (M = B*S = 8192, N = 1024)
    dim3 gg(DIM / 64, (BATCH * SEQ) / 64, 3);
    qkv_gemm<<<gg, 128, 0, stream>>>(xb, wqb, wkb, wvb, bq, bk, bv, qb, kb, vtb);

    // Stage 3: attention (168KB dynamic LDS)
    const int smem = (16 * DIM + 2 * 32 * DIM + 8 * 16 * 32) * (int)sizeof(unsigned short);
    (void)hipFuncSetAttribute((const void*)flash_attn,
                              hipFuncAttributeMaxDynamicSharedMemorySize, smem);
    dim3 ga(SEQ / 16, BATCH);
    flash_attn<<<ga, 256, smem, stream>>>(qb, kb, vtb, out);
}
